// BlockAttnRes_30760555774611
// MI455X (gfx1250) — matrix operand order repaired and checked by reading, not compiled, not run on hardware
//
#include <hip/hip_runtime.h>
#include <hip/hip_bf16.h>

// ---------------------------------------------------------------------------
// BlockAttnRes for MI455X (gfx1250): one workgroup per token, single HBM pass.
//   X (25 x 2048 bf16, LDS)  = [embedding, f_0..f_23]
//   Phase B: Y = X * [X^T | WQ^T]  via v_wmma_f32_16x16x32_bf16 (Gram + dots)
//   Phase C: per-layer RMS-scaled scores -> softmax -> coeff matrix W (24x25)
//   Phase D: H = W * X            via v_wmma_f32_16x16x32_bf16 -> global out
// Traffic ~822 MB total => ~35us floor @ 23.3 TB/s. Compute ~15 GFLOP (noise).
// ---------------------------------------------------------------------------

typedef float        v8f   __attribute__((ext_vector_type(8)));
typedef __bf16       v16bf __attribute__((ext_vector_type(16)));
typedef unsigned int v8u   __attribute__((ext_vector_type(8)));
typedef unsigned short u16;

#define LNUM 24
#define DNUM 2048
#define BNUM 2
#define TNUM 1024
#define NSRC 25          // embedding + 24 layer outputs
#define XS   2056        // padded row stride (u16 elems) for X / WQ in LDS
#define XTS  40          // padded row stride (u16 elems) for X^T / W in LDS
#define EPSV 1e-6f

#define X_BYTES   (32 * XS * 2)        /* 131584 */
#define U_BYTES   (DNUM * XTS * 2)     /* 163840: WQ (phase B) then X^T (phase D) */
#define Y_BYTES   (32 * 64 * 4)        /*   8192 */
#define W_BYTES   (32 * XTS * 2)       /*   2560 */
#define LDS_TOTAL (X_BYTES + U_BYTES + Y_BYTES + W_BYTES)  /* 306176 < 320KB */

__device__ __forceinline__ u16 f2bf(float f) {  // round-to-nearest-even
    unsigned int u = __float_as_uint(f);
    unsigned int r = u + 0x7FFFu + ((u >> 16) & 1u);
    return (u16)(r >> 16);
}

// Build a v16bf WMMA operand from two 16-byte LDS chunks (ds_load_b128 x2).
__device__ __forceinline__ v16bf mk16(const u16* p0, const u16* p1) {
    uint4 a = *reinterpret_cast<const uint4*>(p0);
    uint4 b = *reinterpret_cast<const uint4*>(p1);
    v8u u;
    u[0] = a.x; u[1] = a.y; u[2] = a.z; u[3] = a.w;
    u[4] = b.x; u[5] = b.y; u[6] = b.z; u[7] = b.w;
    return __builtin_bit_cast(v16bf, u);
}

__global__ void __launch_bounds__(256)
blockattnres_kernel(const float* __restrict__ lo,   // (L,B,T,D) layer_outputs
                    const float* __restrict__ emb,  // (B,T,D)
                    const float* __restrict__ qry,  // (L,D)
                    const float* __restrict__ knw,  // (D,)
                    float* __restrict__ out)        // (L,B,T,D)
{
    extern __shared__ char smem[];
    u16*   Xu = reinterpret_cast<u16*>(smem);                         // X   bf16 [32][XS]
    u16*   Uu = reinterpret_cast<u16*>(smem + X_BYTES);               // WQ then X^T
    float* Yf = reinterpret_cast<float*>(smem + X_BYTES + U_BYTES);   // Y   f32  [32][64]
    u16*   Wu = reinterpret_cast<u16*>(smem + X_BYTES + U_BYTES + Y_BYTES); // W bf16 [32][XTS]

    const int tid  = threadIdx.x;
    const int wave = tid >> 5;
    const int lane = tid & 31;
    const int tok  = blockIdx.x;
    const int bI   = tok / TNUM;
    const int t    = tok % TNUM;

    // ---------------- Phase A: stage X and WQ into LDS (f32 -> bf16) --------
    {
        const size_t tokOff = ((size_t)bI * TNUM + t) * DNUM;
        for (int i = tid; i < NSRC * (DNUM / 4); i += 256) {
            const int row = i / (DNUM / 4);
            const int c4  = i % (DNUM / 4);
            const float4* src = reinterpret_cast<const float4*>(
                row == 0 ? (emb + tokOff)
                         : (lo + (((size_t)(row - 1) * BNUM + bI) * TNUM + t) * DNUM));
            const float4 v = src[c4];
            ushort4 o;
            o.x = f2bf(v.x); o.y = f2bf(v.y); o.z = f2bf(v.z); o.w = f2bf(v.w);
            *reinterpret_cast<ushort4*>(&Xu[(size_t)row * XS + (size_t)c4 * 4]) = o;
        }
        for (int i = tid; i < 7 * XS; i += 256) Xu[25 * XS + i] = 0;  // zero pad rows

        // WQ[l][d] = key_norm_weight[d] * queries[l][d]
        for (int i = tid; i < LNUM * (DNUM / 4); i += 256) {
            const int row = i / (DNUM / 4);
            const int c4  = i % (DNUM / 4);
            const float4 q = reinterpret_cast<const float4*>(qry)[row * (DNUM / 4) + c4];
            const float4 w = reinterpret_cast<const float4*>(knw)[c4];
            ushort4 o;
            o.x = f2bf(q.x * w.x); o.y = f2bf(q.y * w.y);
            o.z = f2bf(q.z * w.z); o.w = f2bf(q.w * w.w);
            *reinterpret_cast<ushort4*>(&Uu[(size_t)row * XS + (size_t)c4 * 4]) = o;
        }
        for (int i = tid; i < 8 * XS; i += 256) Uu[24 * XS + i] = 0;  // zero pad rows
        for (int i = tid; i < 32 * XTS; i += 256) Wu[i] = 0;          // zero coeff matrix
    }
    __syncthreads();

    // ---------------- Phase B: Y = X * [X^T | WQ^T]  (bf16 WMMA, f32 acc) ---
    // 6 active waves, one 16x16 C-tile each:
    //   w0: Gram tile (rows 0-15 x cols 0-15)   w1: Gram tile (16-31 x 16-31)
    //   w2..w5: projection tiles X * WQ^T (2 row-tiles x 2 layer-tiles)
    // Only diagonal Gram tiles are needed: 3-row blocks never cross row 15|16.
    if (wave < 6) {
        int mi, nbase, ycol;
        const u16* bsrc;
        if (wave == 0)      { mi = 0; bsrc = Xu; nbase = 0;  ycol = 0;  }
        else if (wave == 1) { mi = 1; bsrc = Xu; nbase = 16; ycol = 16; }
        else {
            const int p = wave - 2;
            mi = p >> 1;
            const int nl = p & 1;
            bsrc = Uu; nbase = nl * 16; ycol = 32 + nl * 16;
        }
        const int rA   = mi * 16 + (lane & 15);
        const int aoff = (lane < 16) ? 0 : 8;      // A: K=[0-7,16-23] / [8-15,24-31]
        const int rB   = nbase + (lane & 15);
        const int boff = (lane < 16) ? 0 : 8;      // B: K=[0-7,16-23] / [8-15,24-31]

        v8f acc0 = {}, acc1 = {};
        for (int kk = 0; kk < 64; kk += 2) {
            const int d0 = kk * 32, d1 = d0 + 32;
            v16bf a0 = mk16(&Xu[(size_t)rA * XS + d0 + aoff],
                            &Xu[(size_t)rA * XS + d0 + 16 + aoff]);
            v16bf b0 = mk16(&bsrc[(size_t)rB * XS + d0 + boff],
                            &bsrc[(size_t)rB * XS + d0 + boff + 16]);
            acc0 = __builtin_amdgcn_wmma_f32_16x16x32_bf16(false, a0, false, b0,
                                                           (short)0, acc0, false, false);
            v16bf a1 = mk16(&Xu[(size_t)rA * XS + d1 + aoff],
                            &Xu[(size_t)rA * XS + d1 + 16 + aoff]);
            v16bf b1 = mk16(&bsrc[(size_t)rB * XS + d1 + boff],
                            &bsrc[(size_t)rB * XS + d1 + boff + 16]);
            acc1 = __builtin_amdgcn_wmma_f32_16x16x32_bf16(false, a1, false, b1,
                                                           (short)0, acc1, false, false);
        }
        // C layout: VGPR r, lanes 0-15 -> M=r, N=lane; lanes 16-31 -> M=r+8
        for (int r = 0; r < 8; ++r) {
            const int m = mi * 16 + r + ((lane < 16) ? 0 : 8);
            Yf[m * 64 + ycol + (lane & 15)] = acc0[r] + acc1[r];
        }
    }
    __syncthreads();

    // ---------------- Phase C: scores -> softmax -> coeff matrix W ----------
    if (tid < LNUM) {
        const int l  = tid;
        const int bn = l / 3;
        const int ii = l % 3;
        float sc[12];
        int ns = 0;
        // source 0: embedding
        {
            const float dot = Yf[32 + l];
            const float ss  = Yf[0];
            sc[ns++] = dot * rsqrtf(ss * (1.0f / DNUM) + EPSV);
        }
        // completed block representations
        for (int j = 1; j <= bn; ++j) {
            const int base = 1 + 3 * (j - 1);
            float dot = 0.f, ss = 0.f;
            for (int k = 0; k < 3; ++k) {
                dot += Yf[(base + k) * 64 + 32 + l];
                for (int k2 = 0; k2 < 3; ++k2) ss += Yf[(base + k) * 64 + (base + k2)];
            }
            sc[ns++] = dot * rsqrtf(ss * (1.0f / DNUM) + EPSV);
        }
        // in-block partial sum
        if (ii > 0) {
            const int base = 1 + 3 * bn;
            float dot = 0.f, ss = 0.f;
            for (int k = 0; k < ii; ++k) {
                dot += Yf[(base + k) * 64 + 32 + l];
                for (int k2 = 0; k2 < ii; ++k2) ss += Yf[(base + k) * 64 + (base + k2)];
            }
            sc[ns++] = dot * rsqrtf(ss * (1.0f / DNUM) + EPSV);
        }
        float mx = sc[0];
        for (int s = 1; s < ns; ++s) mx = fmaxf(mx, sc[s]);
        float sum = 0.f;
        for (int s = 0; s < ns; ++s) { sc[s] = __expf(sc[s] - mx); sum += sc[s]; }
        const float inv = 1.0f / sum;
        // scatter alpha onto basis columns: h_l = sum_s W[l][s] * X[s]
        u16* wr = &Wu[l * XTS];
        wr[0] = f2bf(sc[0] * inv);
        for (int j = 1; j <= bn; ++j) {
            const u16 a = f2bf(sc[j] * inv);
            const int base = 1 + 3 * (j - 1);
            wr[base] = a; wr[base + 1] = a; wr[base + 2] = a;
        }
        if (ii > 0) {
            const u16 a = f2bf(sc[bn + 1] * inv);
            const int base = 1 + 3 * bn;
            for (int k = 0; k < ii; ++k) wr[base + k] = a;
        }
    }
    // transpose X -> X^T into the (now free) WQ region; cols >= NSRC zeroed so
    // no stale-LDS NaN can poison 0*x terms in phase D.
    for (int i = tid; i < DNUM * 32; i += 256) {
        const int d = i >> 5, s = i & 31;
        Uu[(size_t)d * XTS + s] = (s < NSRC) ? Xu[(size_t)s * XS + d] : (u16)0;
    }
    __syncthreads();

    // ---------------- Phase D: H = W * X  (single K=32 WMMA per d-tile) -----
    {
        const int mi   = wave & 1;                 // layer row-tile (fixed per wave)
        const int rA   = mi * 16 + (lane & 15);
        const int aoff = (lane < 16) ? 0 : 8;
        const v16bf Aop = mk16(&Wu[(size_t)rA * XTS + aoff],
                               &Wu[(size_t)rA * XTS + 16 + aoff]);
        const int boff  = (lane < 16) ? 0 : 8;
        const int nlane = lane & 15;
        const int moff  = (lane < 16) ? 0 : 8;
        for (int di = (wave >> 1); di < DNUM / 16; di += 4) {
            const int rB = di * 16 + nlane;
            const v16bf Bop = mk16(&Uu[(size_t)rB * XTS + boff],
                                   &Uu[(size_t)rB * XTS + boff + 16]);
            v8f c = {};
            c = __builtin_amdgcn_wmma_f32_16x16x32_bf16(false, Aop, false, Bop,
                                                        (short)0, c, false, false);
            const int dcol = di * 16 + nlane;
            for (int r = 0; r < 8; ++r) {
                const int lrow = mi * 16 + r + moff;
                if (lrow < LNUM) {
                    out[(((size_t)lrow * BNUM + bI) * TNUM + t) * DNUM + dcol] = c[r];
                }
            }
        }
    }
}

extern "C" void kernel_launch(void* const* d_in, const int* in_sizes, int n_in,
                              void* d_out, int out_size, void* d_ws, size_t ws_size,
                              hipStream_t stream) {
    (void)in_sizes; (void)n_in; (void)out_size; (void)d_ws; (void)ws_size;
    const float* lo  = (const float*)d_in[0];  // layer_outputs (L,B,T,D)
    const float* emb = (const float*)d_in[1];  // embedding     (B,T,D)
    const float* qry = (const float*)d_in[2];  // queries       (L,D)
    const float* knw = (const float*)d_in[3];  // key_norm_weight (D,)
    float* out = (float*)d_out;

    (void)hipFuncSetAttribute(reinterpret_cast<const void*>(blockattnres_kernel),
                              hipFuncAttributeMaxDynamicSharedMemorySize, LDS_TOTAL);
    blockattnres_kernel<<<BNUM * TNUM, 256, LDS_TOTAL, stream>>>(lo, emb, qry, knw, out);
}
